// ICLLoss_38439957299344
// MI455X (gfx1250) — matrix operand order repaired and checked by reading, not compiled, not run on hardware
//
#include <hip/hip_runtime.h>
#include <hip/hip_bf16.h>
#include <stdint.h>

// ---------------------------------------------------------------------------
// CosineAlign + CE loss, fused for gfx1250 (MI455X).
//   Pass 1: L2-normalize rows of emb1/emb2 -> bf16 copies in workspace.
//   Pass 2: tiled U @ V^T via v_wmma_f32_16x16x32_bf16 with fused online
//           logsumexp (flash-softmax) per row; diag captured in-flight;
//           atomic-add of per-row loss/N into d_out.
// Logits matrix (256 MB) is never materialized; bf16 working set (32 MB)
// is L2-resident on the 192 MB L2.
// ---------------------------------------------------------------------------

typedef __attribute__((ext_vector_type(16))) __bf16    bf16x16;
typedef __attribute__((ext_vector_type(8)))  float     f32x8;
typedef __attribute__((ext_vector_type(4)))  float     f32x4;
typedef __attribute__((ext_vector_type(4)))  uint32_t  u32x4;
typedef __attribute__((ext_vector_type(2)))  uint32_t  u32x2;

#define NROWS 8192
#define DDIM  1024
#define WAVES 8               // waves per workgroup in the GEMM kernel
#define CTILES (NROWS / 16)   // 512 column tiles
#define TPW (CTILES / WAVES)  // 64 column tiles per wave

union FragBF { bf16x16 bf; u32x4 u4[2]; };

__device__ __forceinline__ unsigned short f32_to_bf16_rne(float f) {
    uint32_t u = __float_as_uint(f);
    u += 0x7fffu + ((u >> 16) & 1u);   // round-to-nearest-even
    return (unsigned short)(u >> 16);
}

// --------------------------- Pass 0: zero accumulator ----------------------
__global__ void zero_scalar(float* out) { out[0] = 0.0f; }

// --------------------------- Pass 1: row L2-normalize ----------------------
// One wave per row; 8 waves (256 threads) per block.
__global__ __launch_bounds__(256)
void normalize_rows_bf16(const float* __restrict__ in,
                         unsigned short* __restrict__ out) {
    const int lane = threadIdx.x & 31;
    const int wave = threadIdx.x >> 5;
    const int row  = blockIdx.x * WAVES + wave;

    const float* p = in + (size_t)row * DDIM;
    f32x4 v[8];
    float ss = 0.0f;
#pragma unroll
    for (int i = 0; i < 8; ++i) {
        v[i] = *reinterpret_cast<const f32x4*>(p + (lane + 32 * i) * 4);
#pragma unroll
        for (int j = 0; j < 4; ++j) ss += v[i][j] * v[i][j];
    }
#pragma unroll
    for (int off = 1; off < 32; off <<= 1) ss += __shfl_xor(ss, off, 32);

    const float scale = 1.0f / fmaxf(sqrtf(ss), 1e-12f);

    unsigned short* q = out + (size_t)row * DDIM;
#pragma unroll
    for (int i = 0; i < 8; ++i) {
        u32x2 pk;
        pk[0] = (uint32_t)f32_to_bf16_rne(v[i][0] * scale) |
                ((uint32_t)f32_to_bf16_rne(v[i][1] * scale) << 16);
        pk[1] = (uint32_t)f32_to_bf16_rne(v[i][2] * scale) |
                ((uint32_t)f32_to_bf16_rne(v[i][3] * scale) << 16);
        *reinterpret_cast<u32x2*>(q + (lane + 32 * i) * 4) = pk;
    }
}

// ------------------ Pass 2: fused GEMM + online logsumexp ------------------
// One 16-row block per workgroup; 8 waves split the 512 column tiles.
__global__ __launch_bounds__(256)
void cosine_ce_loss(const unsigned short* __restrict__ U,
                    const unsigned short* __restrict__ V,
                    float* __restrict__ out) {
    __shared__ float red_m[WAVES][16];
    __shared__ float red_s[WAVES][16];
    __shared__ float diag[16];

    const int lane = threadIdx.x & 31;
    const int wave = threadIdx.x >> 5;
    const int n15  = lane & 15;
    const int rsel = lane >> 4;          // half-wave select
    const int rowBase = blockIdx.x * 16;

    // A fragment (16x32 bf16): lane holds row rowBase+n15;
    //   lo half: K k+0..15 contiguous ; hi half: K k+16..31
    const unsigned short* arow = U + (size_t)(rowBase + n15) * DDIM;
    const int ka = rsel * 16;
    // B fragment (32x16 bf16): lane holds col = tile*16+n15;
    //   lo half: K k+0..15 contiguous ; hi half: K k+16..31
    const int kb = rsel * 16;

    float m[8], s[8];
#pragma unroll
    for (int r = 0; r < 8; ++r) { m[r] = -3.0e38f; s[r] = 0.0f; }

    const int t0base = wave * TPW;
    for (int tp = 0; tp < TPW; tp += 2) {
        const int t0 = t0base + tp;
        const int t1 = t0 + 1;
        const unsigned short* b0row = V + (size_t)(t0 * 16 + n15) * DDIM;
        const unsigned short* b1row = V + (size_t)(t1 * 16 + n15) * DDIM;

        f32x8 acc0 = {0, 0, 0, 0, 0, 0, 0, 0};
        f32x8 acc1 = {0, 0, 0, 0, 0, 0, 0, 0};

        for (int k = 0; k < DDIM; k += 32) {
            FragBF a, b0, b1;
            a.u4[0]  = *reinterpret_cast<const u32x4*>(arow  + k + ka);
            a.u4[1]  = *reinterpret_cast<const u32x4*>(arow  + k + ka + 8);
            b0.u4[0] = *reinterpret_cast<const u32x4*>(b0row + k + kb);
            b0.u4[1] = *reinterpret_cast<const u32x4*>(b0row + k + kb + 8);
            b1.u4[0] = *reinterpret_cast<const u32x4*>(b1row + k + kb);
            b1.u4[1] = *reinterpret_cast<const u32x4*>(b1row + k + kb + 8);
            acc0 = __builtin_amdgcn_wmma_f32_16x16x32_bf16(
                false, a.bf, false, b0.bf, (short)0, acc0, false, false);
            acc1 = __builtin_amdgcn_wmma_f32_16x16x32_bf16(
                false, a.bf, false, b1.bf, (short)0, acc1, false, false);
        }

        // Online logsumexp update: each lane sees 1 column per row per tile.
#pragma unroll
        for (int r = 0; r < 8; ++r) {
            float x = acc0[r];
            float d = x - m[r];
            float e = __expf(-fabsf(d));
            if (d > 0.0f) { s[r] = s[r] * e + 1.0f; m[r] = x; }
            else          { s[r] += e; }
            x = acc1[r];
            d = x - m[r];
            e = __expf(-fabsf(d));
            if (d > 0.0f) { s[r] = s[r] * e + 1.0f; m[r] = x; }
            else          { s[r] += e; }
        }

        // Diagonal tile: column tile index == row-block index.
        if (t0 == (int)blockIdx.x || t1 == (int)blockIdx.x) {
            const bool first = (t0 == (int)blockIdx.x);
            float tmp[8];
#pragma unroll
            for (int r = 0; r < 8; ++r) tmp[r] = first ? acc0[r] : acc1[r];
            // C/D layout: lanes 0-15 vgpr r -> M=r; lanes 16-31 vgpr r -> M=r+8.
            // Diagonal (M == n15) lives in lanes 0..7 (r=n15) and 24..31 (r=n15-8).
            const int r = n15 - 8 * rsel;
            if (r >= 0 && r < 8) diag[n15] = tmp[r];
        }
    }

    // Merge (m,s) across the 16 lanes of each half-wave.
#pragma unroll
    for (int r = 0; r < 8; ++r) {
        float mm = m[r], sv = s[r];
#pragma unroll
        for (int off = 1; off < 16; off <<= 1) {
            float mo = __shfl_xor(mm, off, 32);
            float so = __shfl_xor(sv, off, 32);
            float M2 = fmaxf(mm, mo);
            sv = sv * __expf(mm - M2) + so * __expf(mo - M2);
            mm = M2;
        }
        m[r] = mm; s[r] = sv;
    }

    if (n15 == 0) {
#pragma unroll
        for (int r = 0; r < 8; ++r) {
            red_m[wave][rsel * 8 + r] = m[r];
            red_s[wave][rsel * 8 + r] = s[r];
        }
    }
    __syncthreads();

    // Merge across waves; emit per-row loss contribution.
    if (threadIdx.x < 16) {
        const int row = threadIdx.x;
        float mm = red_m[0][row], sv = red_s[0][row];
#pragma unroll
        for (int w = 1; w < WAVES; ++w) {
            float mo = red_m[w][row], so = red_s[w][row];
            float M2 = fmaxf(mm, mo);
            sv = sv * __expf(mm - M2) + so * __expf(mo - M2);
            mm = M2;
        }
        const float loss_i = (mm + __logf(sv)) - diag[row];
        atomicAdd(out, loss_i * (1.0f / (float)NROWS));
    }
}

// ---------------------------------------------------------------------------
extern "C" void kernel_launch(void* const* d_in, const int* in_sizes, int n_in,
                              void* d_out, int out_size, void* d_ws, size_t ws_size,
                              hipStream_t stream) {
    (void)in_sizes; (void)n_in; (void)out_size; (void)ws_size;
    const float* emb1 = (const float*)d_in[0];
    const float* emb2 = (const float*)d_in[1];
    float* out = (float*)d_out;

    unsigned short* Ubf = (unsigned short*)d_ws;                 // 16 MiB
    unsigned short* Vbf = Ubf + (size_t)NROWS * DDIM;            // 16 MiB

    zero_scalar<<<1, 1, 0, stream>>>(out);
    normalize_rows_bf16<<<NROWS / WAVES, 256, 0, stream>>>(emb1, Ubf);
    normalize_rows_bf16<<<NROWS / WAVES, 256, 0, stream>>>(emb2, Vbf);
    cosine_ce_loss<<<NROWS / 16, 256, 0, stream>>>(Ubf, Vbf, out);
}
